// EncoderLayer_64682207478416
// MI455X (gfx1250) — matrix operand order repaired and checked by reading, not compiled, not run on hardware
//
#include <hip/hip_runtime.h>
#include <hip/hip_bf16.h>
#include <math.h>

typedef unsigned short u16;
typedef __attribute__((ext_vector_type(16))) __bf16 v16bf;
typedef __attribute__((ext_vector_type(8)))  float  v8f;

#define B_   4
#define S_   2048
#define D_   512
#define H_   8
#define E_   64
#define F_   2048
#define BS_  (B_*S_)
#define BH_  (B_*H_)
#define SCALE_ 0.125f  /* 1/sqrt(64) */

__device__ __forceinline__ u16 f2bf(float f) {
  unsigned u = __float_as_uint(f);
  u += 0x7FFFu + ((u >> 16) & 1u);     // round-to-nearest-even
  return (u16)(u >> 16);
}
__device__ __forceinline__ float gelu_exact(float v) {
  return 0.5f * v * (1.0f + erff(v * 0.70710678118654752f));
}

union FragU { v16bf v; uint4 q[2]; };

// A-fragment (16x32 bf16): lane L -> M = L&15; j<8: K=j+8*hl; j>=8: K=j+8+8*hl
__device__ __forceinline__ v16bf load_a_frag(const u16* __restrict__ A, int lda,
                                             int row, int kbase, int lane) {
  int hl = lane >> 4;
  const u16* p = A + (size_t)(row + (lane & 15)) * lda + kbase + 8 * hl;
  FragU f;
  f.q[0] = *(const uint4*)p;           // K = kbase+8hl .. +7
  f.q[1] = *(const uint4*)(p + 16);    // K = kbase+16+8hl .. +7
  return f.v;
}
// B-fragment (32x16 bf16) from B^T storage [N][K]: lane L -> N=L&15; j<8: K=kbase+j+8*hl; j>=8: K=kbase+j+8+8*hl
__device__ __forceinline__ v16bf load_b_frag(const u16* __restrict__ BT, int ldb,
                                             int nbase, int kbase, int lane) {
  int hl = lane >> 4;
  const u16* p = BT + (size_t)(nbase + (lane & 15)) * ldb + kbase + 8 * hl;
  FragU f;
  f.q[0] = *(const uint4*)p;
  f.q[1] = *(const uint4*)(p + 16);
  return f.v;
}
// A-fragment from LDS P-tile, row stride 40 u16 (80B, conflict-spread, 16B aligned)
__device__ __forceinline__ v16bf load_a_frag_lds(const u16* p0, int lane) {
  int hl = lane >> 4;
  const u16* p = p0 + (lane & 15) * 40 + 8 * hl;
  FragU f;
  f.q[0] = *(const uint4*)p;
  f.q[1] = *(const uint4*)(p + 16);
  return f.v;
}
__device__ __forceinline__ v8f wmma_bf16(v16bf a, v16bf b, v8f c) {
  return __builtin_amdgcn_wmma_f32_16x16x32_bf16(false, a, false, b, (short)0, c,
                                                 false, false);
}

/* ---------------- LayerNorm (ddof=1, /(std+eps)) -> bf16 ---------------- */
__global__ __launch_bounds__(256) void ln_kernel(const float* __restrict__ X,
                                                 const float* __restrict__ gamma,
                                                 const float* __restrict__ beta,
                                                 u16* __restrict__ out) {
  int row = blockIdx.x;
  const float* x = X + (size_t)row * D_;
  int t = threadIdx.x;
  float v0 = x[t], v1 = x[t + 256];
  float s = v0 + v1, ss = v0 * v0 + v1 * v1;
  __shared__ float sh[16];
  int lane = t & 31, wave = t >> 5;
  #pragma unroll
  for (int d = 1; d < 32; d <<= 1) { s += __shfl_xor(s, d, 32); ss += __shfl_xor(ss, d, 32); }
  if (lane == 0) { sh[wave] = s; sh[8 + wave] = ss; }
  __syncthreads();
  if (t == 0) {
    float S = 0.f, SS = 0.f;
    for (int i = 0; i < 8; ++i) { S += sh[i]; SS += sh[8 + i]; }
    float mean = S * (1.0f / D_);
    float var  = (SS - (float)D_ * mean * mean) * (1.0f / (D_ - 1));
    sh[0] = mean;
    sh[1] = 1.0f / (sqrtf(fmaxf(var, 0.f)) + 1e-10f);
  }
  __syncthreads();
  float mean = sh[0], k = sh[1];
  out[(size_t)row * D_ + t]       = f2bf(gamma[t]       * (v0 - mean) * k + beta[t]);
  out[(size_t)row * D_ + t + 256] = f2bf(gamma[t + 256] * (v1 - mean) * k + beta[t + 256]);
}

/* -------- fp32 [batch][R][C] -> bf16 [batch][C][R] (weight prep) -------- */
__global__ __launch_bounds__(256) void transpose_bf16_kernel(const float* __restrict__ in,
                                                             u16* __restrict__ out,
                                                             int R, int C) {
  size_t base = (size_t)blockIdx.y * R * C;
  int i = blockIdx.x * 256 + threadIdx.x;
  if (i < R * C) {
    int r = i / C, c = i % C;
    out[base + (size_t)c * R + r] = f2bf(in[base + i]);
  }
}

/* ---------------- QKV projections (SCALE folded into Q) ---------------- */
__global__ __launch_bounds__(256) void qkv_kernel(
    const u16* __restrict__ xn,                       // [BS, D] bf16
    const u16* __restrict__ WqT, const u16* __restrict__ WkT, const u16* __restrict__ WvT,
    const float* __restrict__ bq, const float* __restrict__ bk, const float* __restrict__ bv,
    u16* __restrict__ Q, u16* __restrict__ Kmat, u16* __restrict__ VT) {
  int lane = threadIdx.x & 31, wave = threadIdx.x >> 5;
  int idx = blockIdx.x;
  int st = idx % (S_ / 128); idx /= (S_ / 128);
  int bh = idx % BH_;        idx /= BH_;
  int mat = idx;                                      // 0=Q 1=K 2=V
  int b = bh / H_, h = bh % H_;
  const u16*  W    = (mat == 0 ? WqT : mat == 1 ? WkT : WvT) + (size_t)h * E_ * D_;
  const float* bia = (mat == 0 ? bq  : mat == 1 ? bk  : bv)  + h * E_;
  int mrow = b * S_ + st * 128 + wave * 16;
  v8f acc[4] = {};
  for (int k = 0; k < D_; k += 32) {
    v16bf a = load_a_frag(xn, D_, mrow, k, lane);
    #pragma unroll
    for (int t = 0; t < 4; ++t)
      acc[t] = wmma_bf16(a, load_b_frag(W, D_, 16 * t, k, lane), acc[t]);
  }
  int hl = lane >> 4;
  int sloc = st * 128 + wave * 16 + 8 * hl;           // first of 8 consecutive s rows
  if (mat == 2) {                                     // V stored transposed: VT[bh][e][s]
    #pragma unroll
    for (int t = 0; t < 4; ++t) {
      int e = 16 * t + (lane & 15);
      float bv_ = bia[e];
      __align__(16) u16 tmp[8];
      #pragma unroll
      for (int r = 0; r < 8; ++r) tmp[r] = f2bf(acc[t][r] + bv_);
      *(uint4*)(VT + ((size_t)bh * E_ + e) * S_ + sloc) = *(const uint4*)tmp;
    }
  } else {
    u16* Om = (mat == 0 ? Q : Kmat) + (size_t)bh * S_ * E_;
    float scl = (mat == 0) ? SCALE_ : 1.0f;
    #pragma unroll
    for (int t = 0; t < 4; ++t) {
      int e = 16 * t + (lane & 15);
      float bv_ = bia[e];
      #pragma unroll
      for (int r = 0; r < 8; ++r)
        Om[(size_t)(sloc + r) * E_ + e] = f2bf((acc[t][r] + bv_) * scl);
    }
  }
}

/* -------------- flash attention: one (bh, 64-row q-tile) per block ------ */
__global__ __launch_bounds__(128) void attn_kernel(
    const u16* __restrict__ Q, const u16* __restrict__ Kmat,
    const u16* __restrict__ VT, u16* __restrict__ attout) {
  __shared__ __attribute__((aligned(16))) u16 plds[4 * 16 * 40];
  int lane = threadIdx.x & 31, wave = threadIdx.x >> 5;
  int qtiles = S_ / 64;
  int qt = blockIdx.x % qtiles, bh = blockIdx.x / qtiles;
  int qrow = qt * 64 + wave * 16;
  const u16* Qm = Q    + (size_t)bh * S_ * E_;
  const u16* Km = Kmat + (size_t)bh * S_ * E_;
  const u16* Vm = VT   + (size_t)bh * E_ * S_;
  u16* wl = plds + wave * 640;
  int hl = lane >> 4;

  v16bf qa0 = load_a_frag(Qm, E_, qrow, 0, lane);     // Q reused for all keys
  v16bf qa1 = load_a_frag(Qm, E_, qrow, 32, lane);
  v8f o[4] = {};
  float mrun[8], lrun[8];
  #pragma unroll
  for (int r = 0; r < 8; ++r) { mrun[r] = -1e30f; lrun[r] = 0.f; }

  for (int kb = 0; kb < S_; kb += 32) {
    v8f s0 = {}, s1 = {};
    s0 = wmma_bf16(qa0, load_b_frag(Km, E_, kb,      0,  lane), s0);
    s0 = wmma_bf16(qa1, load_b_frag(Km, E_, kb,      32, lane), s0);
    s1 = wmma_bf16(qa0, load_b_frag(Km, E_, kb + 16, 0,  lane), s1);
    s1 = wmma_bf16(qa1, load_b_frag(Km, E_, kb + 16, 32, lane), s1);
    #pragma unroll
    for (int r = 0; r < 8; ++r) {
      float v0 = s0[r], v1 = s1[r];
      float mx = fmaxf(v0, v1);
      #pragma unroll
      for (int d = 1; d < 16; d <<= 1) mx = fmaxf(mx, __shfl_xor(mx, d, 32));
      float mnew  = fmaxf(mrun[r], mx);
      float alpha = __expf(mrun[r] - mnew);
      float p0 = __expf(v0 - mnew), p1 = __expf(v1 - mnew);
      float rs = p0 + p1;
      #pragma unroll
      for (int d = 1; d < 16; d <<= 1) rs += __shfl_xor(rs, d, 32);
      lrun[r] = lrun[r] * alpha + rs;
      mrun[r] = mnew;
      s0[r] = p0; s1[r] = p1;
      #pragma unroll
      for (int t = 0; t < 4; ++t) o[t][r] = o[t][r] * alpha;
    }
    // transpose P (C-frag layout) through LDS to A-frag layout (bf16)
    #pragma unroll
    for (int r = 0; r < 8; ++r) {
      int row = 8 * hl + r;
      wl[row * 40 + (lane & 15)]      = f2bf(s0[r]);
      wl[row * 40 + 16 + (lane & 15)] = f2bf(s1[r]);
    }
    v16bf pa = load_a_frag_lds(wl, lane);             // same-wave DS ops are in-order
    #pragma unroll
    for (int t = 0; t < 4; ++t)
      o[t] = wmma_bf16(pa, load_b_frag(Vm, S_, 16 * t, kb, lane), o[t]);
  }
  int b = bh / H_, h = bh % H_;
  #pragma unroll
  for (int t = 0; t < 4; ++t) {
    int e = 16 * t + (lane & 15);
    #pragma unroll
    for (int r = 0; r < 8; ++r) {
      int s = qrow + 8 * hl + r;
      attout[((size_t)(b * S_ + s)) * (H_ * E_) + h * E_ + e] = f2bf(o[t][r] / lrun[r]);
    }
  }
}

/* ------------- generic bf16 WMMA GEMM: C = A * B^T + bias, epilogue ----- */
enum { EPI_BF16 = 0, EPI_GELU_BF16 = 1, EPI_RES_F32 = 2, EPI_GELU_RES_F32 = 3 };

template <int EPI>
__global__ __launch_bounds__(256) void gemm_bf16_kernel(
    const u16* __restrict__ A, int lda, const u16* __restrict__ BT, int ldb,
    const float* __restrict__ bias, const float* __restrict__ Res,
    void* __restrict__ Out, int ldc, int N) {
  int lane = threadIdx.x & 31, wave = threadIdx.x >> 5;
  int nblocks = N / 64;
  int bm = (blockIdx.x / nblocks) * 128;
  int bn = (blockIdx.x % nblocks) * 64;
  int mrow = bm + wave * 16;
  int K = ldb;                                        // B^T is [N][K] dense
  v8f acc[4] = {};
  for (int k = 0; k < K; k += 32) {
    v16bf a = load_a_frag(A, lda, mrow, k, lane);     // reused across 4 n-tiles
    #pragma unroll
    for (int t = 0; t < 4; ++t)
      acc[t] = wmma_bf16(a, load_b_frag(BT, ldb, bn + 16 * t, k, lane), acc[t]);
  }
  int hl = lane >> 4;
  #pragma unroll
  for (int t = 0; t < 4; ++t) {
    int n = bn + 16 * t + (lane & 15);
    float bv = bias[n];
    #pragma unroll
    for (int r = 0; r < 8; ++r) {
      int m = mrow + 8 * hl + r;
      float v = acc[t][r] + bv;
      if (EPI == EPI_GELU_BF16 || EPI == EPI_GELU_RES_F32) v = gelu_exact(v);
      if (EPI == EPI_RES_F32 || EPI == EPI_GELU_RES_F32)
        ((float*)Out)[(size_t)m * ldc + n] = v + Res[(size_t)m * ldc + n];
      else
        ((u16*)Out)[(size_t)m * ldc + n] = f2bf(v);
    }
  }
}

/* ------------------------------- launcher ------------------------------- */
extern "C" void kernel_launch(void* const* d_in, const int* in_sizes, int n_in,
                              void* d_out, int out_size, void* d_ws, size_t ws_size,
                              hipStream_t stream) {
  (void)in_sizes; (void)n_in; (void)out_size; (void)ws_size;
  const float* x      = (const float*)d_in[0];
  const float* Wq     = (const float*)d_in[1];
  const float* bq     = (const float*)d_in[2];
  const float* Wk     = (const float*)d_in[3];
  const float* bk     = (const float*)d_in[4];
  const float* Wv     = (const float*)d_in[5];
  const float* bv     = (const float*)d_in[6];
  const float* Wp     = (const float*)d_in[7];
  const float* bp     = (const float*)d_in[8];
  const float* gamma1 = (const float*)d_in[9];
  const float* beta1  = (const float*)d_in[10];
  const float* gamma2 = (const float*)d_in[11];
  const float* beta2  = (const float*)d_in[12];
  const float* W1     = (const float*)d_in[13];
  const float* b1     = (const float*)d_in[14];
  const float* W2     = (const float*)d_in[15];
  const float* b2     = (const float*)d_in[16];
  float* out = (float*)d_out;

  size_t off = 0;
  auto wsalloc = [&](size_t bytes) -> void* {
    void* p = (char*)d_ws + off;
    off += (bytes + 255) & ~(size_t)255;
    return p;
  };
  u16*  xn   = (u16*)wsalloc((size_t)BS_ * D_ * 2);
  u16*  WqT  = (u16*)wsalloc((size_t)H_ * E_ * D_ * 2);
  u16*  WkT  = (u16*)wsalloc((size_t)H_ * E_ * D_ * 2);
  u16*  WvT  = (u16*)wsalloc((size_t)H_ * E_ * D_ * 2);
  u16*  WpT  = (u16*)wsalloc((size_t)D_ * D_ * 2);
  u16*  W1T  = (u16*)wsalloc((size_t)F_ * D_ * 2);
  u16*  W2T  = (u16*)wsalloc((size_t)D_ * F_ * 2);
  u16*  Qb   = (u16*)wsalloc((size_t)BH_ * S_ * E_ * 2);
  u16*  Kb   = (u16*)wsalloc((size_t)BH_ * S_ * E_ * 2);
  u16*  VTb  = (u16*)wsalloc((size_t)BH_ * E_ * S_ * 2);
  u16*  attb = (u16*)wsalloc((size_t)BS_ * D_ * 2);
  float* x1  = (float*)wsalloc((size_t)BS_ * D_ * 4);
  u16*  x1n  = (u16*)wsalloc((size_t)BS_ * D_ * 2);
  u16*  hb   = (u16*)wsalloc((size_t)BS_ * F_ * 2);

  // 1) LN1 -> bf16
  ln_kernel<<<BS_, 256, 0, stream>>>(x, gamma1, beta1, xn);
  // 2) weight prep: fp32 -> bf16, transposed for contiguous B-fragments
  transpose_bf16_kernel<<<dim3((D_ * E_ + 255) / 256, H_), 256, 0, stream>>>(Wq, WqT, D_, E_);
  transpose_bf16_kernel<<<dim3((D_ * E_ + 255) / 256, H_), 256, 0, stream>>>(Wk, WkT, D_, E_);
  transpose_bf16_kernel<<<dim3((D_ * E_ + 255) / 256, H_), 256, 0, stream>>>(Wv, WvT, D_, E_);
  transpose_bf16_kernel<<<dim3((D_ * D_ + 255) / 256, 1), 256, 0, stream>>>(Wp, WpT, H_ * E_, D_);
  transpose_bf16_kernel<<<dim3((D_ * F_ + 255) / 256, 1), 256, 0, stream>>>(W1, W1T, D_, F_);
  transpose_bf16_kernel<<<dim3((D_ * F_ + 255) / 256, 1), 256, 0, stream>>>(W2, W2T, F_, D_);
  // 3) QKV projections (V stored transposed)
  qkv_kernel<<<3 * BH_ * (S_ / 128), 256, 0, stream>>>(xn, WqT, WkT, WvT, bq, bk, bv,
                                                       Qb, Kb, VTb);
  // 4) flash attention
  attn_kernel<<<BH_ * (S_ / 64), 128, 0, stream>>>(Qb, Kb, VTb, attb);
  // 5) output projection + residual (fp32 x1)
  gemm_bf16_kernel<EPI_RES_F32><<<(BS_ / 128) * (D_ / 64), 256, 0, stream>>>(
      attb, D_, WpT, D_, bp, x, (void*)x1, D_, D_);
  // 6) LN2 -> bf16
  ln_kernel<<<BS_, 256, 0, stream>>>(x1, gamma2, beta2, x1n);
  // 7) FFN1: GELU -> bf16
  gemm_bf16_kernel<EPI_GELU_BF16><<<(BS_ / 128) * (F_ / 64), 256, 0, stream>>>(
      x1n, D_, W1T, D_, b1, nullptr, (void*)hb, F_, F_);
  // 8) FFN2: GELU + residual -> fp32 d_out
  gemm_bf16_kernel<EPI_GELU_RES_F32><<<(BS_ / 128) * (D_ / 64), 256, 0, stream>>>(
      hb, F_, W2T, F_, b2, x1, (void*)out, D_, D_);
}
